// MyRNN_9156870275572
// MI455X (gfx1250) — compile-verified, matrix operand order checked by reading, not run on hardware
//
#include <hip/hip_runtime.h>
#include <stdint.h>

// ---------------------------------------------------------------------------
// Types for CDNA5 WMMA (wave32): v16bf A/B fragments, v8f C/D accumulator.
// ---------------------------------------------------------------------------
typedef __attribute__((ext_vector_type(16))) __bf16 v16bf;
typedef __attribute__((ext_vector_type(8)))  __bf16 v8bf;
typedef __attribute__((ext_vector_type(4)))  __bf16 v4bf;
typedef __attribute__((ext_vector_type(8)))  float  v8f;
typedef __attribute__((ext_vector_type(4)))  float  v4f;

#define LDS_STRIDE 40   // 32 payload + 8 pad; rows stay 16B aligned (80B pitch)
#define USE_ASYNC_B 1   // stage B tiles with global_load_async_to_lds_b128

#if USE_ASYNC_B
#define ASYNC_WAIT() asm volatile("s_wait_asynccnt 0" ::: "memory")
#else
#define ASYNC_WAIT()
#endif

// Per-lane fragment load from an LDS row holding one 32-wide K-chunk.
// ISA 16-bit A layout: lanes 0-15 -> K {0..7,16..23}, lanes 16-31 -> K {8..15,24..31}.
// B fragments use the identical pattern on rows of the pre-transposed weights.
__device__ __forceinline__ v16bf frag_from_lds(const __bf16* rowp, int half) {
  v8bf lo = *(const v8bf*)(rowp + half * 8);
  v8bf hi = *(const v8bf*)(rowp + 16 + half * 8);
  return __builtin_shufflevector(lo, hi, 0, 1, 2, 3, 4, 5, 6, 7,
                                         8, 9, 10, 11, 12, 13, 14, 15);
}

// Stage this thread's 16 f32 of the A tile -> bf16 in LDS (2x ds_store_b128).
__device__ __forceinline__ void stage_A(const float* __restrict__ ap,
                                        __bf16* __restrict__ ad) {
  v4f f0 = *(const v4f*)(ap);
  v4f f1 = *(const v4f*)(ap + 4);
  v4f f2 = *(const v4f*)(ap + 8);
  v4f f3 = *(const v4f*)(ap + 12);
  v8bf b0, b1;
#pragma unroll
  for (int i = 0; i < 4; ++i) {
    b0[i]     = (__bf16)f0[i];
    b0[4 + i] = (__bf16)f1[i];
    b1[i]     = (__bf16)f2[i];
    b1[4 + i] = (__bf16)f3[i];
  }
  *(v8bf*)(ad)     = b0;
  *(v8bf*)(ad + 8) = b1;
}

// Stage this thread's 16 bf16 of the (pre-transposed) W tile into LDS.
// Async path: ISA adds IOFFSET to BOTH the LDS and global addresses, so one
// operand pair with offset:16 covers the second 16 bytes. Tracked by ASYNCcnt.
__device__ __forceinline__ void stage_B(const __bf16* __restrict__ wp,
                                        __bf16* __restrict__ bd) {
#if USE_ASYNC_B
  unsigned int l0 = (unsigned int)(uintptr_t)bd;  // low 32 bits = LDS offset
  asm volatile("global_load_async_to_lds_b128 %0, %1, off"
               :: "v"(l0), "v"(wp) : "memory");
  asm volatile("global_load_async_to_lds_b128 %0, %1, off offset:16"
               :: "v"(l0), "v"(wp) : "memory");
#else
  v8bf w0 = *(const v8bf*)(wp);
  v8bf w1 = *(const v8bf*)(wp + 8);
  *(v8bf*)(bd)     = w0;
  *(v8bf*)(bd + 8) = w1;
#endif
}

// Accumulate C(64x64 block tile) += A(f32->bf16) * Wt^T, double-buffered LDS.
// A: M x K row-major (lda). Wt: N x K row-major bf16 (pre-transposed weights).
// Block = 128 threads, 4 waves in 2x2; each wave owns 32x32 = 2x2 wmma tiles.
__device__ __forceinline__ void gemm_accum_bf16(
    const float* __restrict__ A, int lda, int K,
    const __bf16* __restrict__ Wt,
    __bf16 (*Asb)[LDS_STRIDE], __bf16 (*Bsb)[LDS_STRIDE],
    int bm, int bn, int tid, int lane, int wm, int wn,
    v8f acc[2][2])
{
  const int srow = tid >> 1;         // 0..63 : tile row (M for A, N for B)
  const int scol = (tid & 1) * 16;   // 0/16  : K offset within 32-chunk

  const float*  aptr = A  + (size_t)(bm * 64 + srow) * lda + scol;
  const __bf16* wptr = Wt + (size_t)(bn * 64 + srow) * K   + scol;

  const int half = lane >> 4;
  const int rA   = wm * 32 + (lane & 15);
  const int cB   = wn * 32 + (lane & 15);
  const int nk   = K >> 5;

  // Preload chunk 0 into buffer 0.
  stage_A(aptr, &Asb[srow][scol]);
  stage_B(wptr, &Bsb[srow][scol]);
  ASYNC_WAIT();
  __syncthreads();

  for (int i = 0; i < nk; ++i) {
    const int p = i & 1;
    if (i + 1 < nk) {
      const int q = p ^ 1;
      stage_A(aptr + (i + 1) * 32, &Asb[q * 64 + srow][scol]);
      stage_B(wptr + (i + 1) * 32, &Bsb[q * 64 + srow][scol]);
      if (i + 2 < nk) {
        __builtin_prefetch(aptr + (i + 2) * 32, 0, 1);  // global_prefetch_b8
        __builtin_prefetch(wptr + (i + 2) * 32, 0, 1);
      }
    }

    v16bf a_lo = frag_from_lds(&Asb[p * 64 + rA][0],      half);
    v16bf a_hi = frag_from_lds(&Asb[p * 64 + rA + 16][0], half);
    v16bf b_lo = frag_from_lds(&Bsb[p * 64 + cB][0],      half);
    v16bf b_hi = frag_from_lds(&Bsb[p * 64 + cB + 16][0], half);

    acc[0][0] = __builtin_amdgcn_wmma_f32_16x16x32_bf16(
        false, a_lo, false, b_lo, (short)0, acc[0][0], false, false);
    acc[0][1] = __builtin_amdgcn_wmma_f32_16x16x32_bf16(
        false, a_lo, false, b_hi, (short)0, acc[0][1], false, false);
    acc[1][0] = __builtin_amdgcn_wmma_f32_16x16x32_bf16(
        false, a_hi, false, b_lo, (short)0, acc[1][0], false, false);
    acc[1][1] = __builtin_amdgcn_wmma_f32_16x16x32_bf16(
        false, a_hi, false, b_hi, (short)0, acc[1][1], false, false);

    ASYNC_WAIT();
    __syncthreads();
  }
}

// C[64x64 tile] = maybe_relu( A1*W1t^T (+ A2*W2t^T) + bias ), C f32, M=128.
__global__ void __launch_bounds__(128)
rnn_gemm_bf16_kernel(const float* __restrict__ A1, int lda1, int K1,
                     const __bf16* __restrict__ W1t,
                     const float* __restrict__ A2, int lda2, int K2,
                     const __bf16* __restrict__ W2t,
                     const float* __restrict__ bias,
                     float* __restrict__ C, int ldc,
                     int relu_flag)
{
  __shared__ __bf16 As[2 * 64][LDS_STRIDE];  // double-buffered
  __shared__ __bf16 Bs[2 * 64][LDS_STRIDE];

  const int bm   = blockIdx.x;
  const int bn   = blockIdx.y;
  const int tid  = threadIdx.x;
  const int lane = tid & 31;
  const int wave = tid >> 5;
  const int wm   = wave & 1;
  const int wn   = wave >> 1;

  v8f acc[2][2] = {};

  gemm_accum_bf16(A1, lda1, K1, W1t, As, Bs, bm, bn, tid, lane, wm, wn, acc);
  if (A2) {
    gemm_accum_bf16(A2, lda2, K2, W2t, As, Bs, bm, bn, tid, lane, wm, wn, acc);
  }

  // C/D layout: VGPR v -> row (v + 8*(lane>=16)), col = lane&15 per 16x16 tile.
  const int half = lane >> 4;
#pragma unroll
  for (int i = 0; i < 2; ++i) {
#pragma unroll
    for (int j = 0; j < 2; ++j) {
      const int col = bn * 64 + wn * 32 + j * 16 + (lane & 15);
      const float bv = bias[col];
#pragma unroll
      for (int v = 0; v < 8; ++v) {
        const int row = bm * 64 + wm * 32 + i * 16 + half * 8 + v;
        float val = acc[i][j][v] + bv;
        if (relu_flag) val = fmaxf(val, 0.0f);
        C[(size_t)row * ldc + col] = val;
      }
    }
  }
}

// One-time: W (K x N, f32) -> Wt (N x K, bf16), LDS-tiled 32x32 transpose.
__global__ void __launch_bounds__(256)
transpose_to_bf16_kernel(const float* __restrict__ W, __bf16* __restrict__ Wt,
                         int K, int N)
{
  __shared__ __bf16 tile[32][33];
  const int kb = blockIdx.x * 32;
  const int nb = blockIdx.y * 32;
  const int r  = threadIdx.x >> 3;        // 0..31
  const int c  = (threadIdx.x & 7) * 4;   // 0..28

  v4f f = *(const v4f*)(W + (size_t)(kb + r) * N + nb + c);
#pragma unroll
  for (int i = 0; i < 4; ++i) tile[c + i][r] = (__bf16)f[i];
  __syncthreads();

  v4bf o;
#pragma unroll
  for (int i = 0; i < 4; ++i) o[i] = tile[r][c + i];
  *(v4bf*)(Wt + (size_t)(nb + r) * K + kb + c) = o;
}

extern "C" void kernel_launch(void* const* d_in, const int* in_sizes, int n_in,
                              void* d_out, int out_size, void* d_ws, size_t ws_size,
                              hipStream_t stream) {
  (void)in_sizes; (void)n_in; (void)out_size; (void)ws_size;
  const int T = 256, B = 128, D = 1024, H = 2048, Q = 1024;

  const float* x    = (const float*)d_in[0];  // (T,B,D)
  const float* h0   = (const float*)d_in[1];  // (B,H)
  const float* W_xh = (const float*)d_in[2];  // (D,H)
  const float* W_hh = (const float*)d_in[3];  // (H,H)
  const float* b_h  = (const float*)d_in[4];  // (H,)
  const float* W_hq = (const float*)d_in[5];  // (H,Q)
  const float* b_q  = (const float*)d_in[6];  // (Q,)

  float* out   = (float*)d_out;               // (T*B, Q)
  float* h_out = out + (size_t)T * B * Q;     // (B, H) final hidden state

  // Workspace: transposed bf16 weights + f32 hidden ping-pong (~18 MB).
  char* ws = (char*)d_ws;
  size_t off = 0;
  __bf16* WxhT = (__bf16*)(ws + off); off += (size_t)D * H * sizeof(__bf16); // (H,D)
  __bf16* WhhT = (__bf16*)(ws + off); off += (size_t)H * H * sizeof(__bf16); // (H,H)
  __bf16* WhqT = (__bf16*)(ws + off); off += (size_t)H * Q * sizeof(__bf16); // (Q,H)
  float*  hbuf0 = (float*)(ws + off); off += (size_t)B * H * sizeof(float);
  float*  hbuf1 = (float*)(ws + off);

  // One-time (per call) weight transpose + bf16 conversion.
  transpose_to_bf16_kernel<<<dim3(D / 32, H / 32), 256, 0, stream>>>(W_xh, WxhT, D, H);
  transpose_to_bf16_kernel<<<dim3(H / 32, H / 32), 256, 0, stream>>>(W_hh, WhhT, H, H);
  transpose_to_bf16_kernel<<<dim3(H / 32, Q / 32), 256, 0, stream>>>(W_hq, WhqT, H, Q);
  hipMemcpyAsync(hbuf0, h0, (size_t)B * H * sizeof(float),
                 hipMemcpyDeviceToDevice, stream);

  float* hcur  = hbuf0;
  float* hnext = hbuf1;
  const dim3 grid_h(B / 64, H / 64);  // (2, 32)
  const dim3 grid_q(B / 64, Q / 64);  // (2, 16)

  for (int t = 0; t < T; ++t) {
    // h_next = relu(x_t @ W_xh + h @ W_hh + b_h)
    rnn_gemm_bf16_kernel<<<grid_h, 128, 0, stream>>>(
        x + (size_t)t * B * D, D, D, WxhT,
        hcur, H, H, WhhT,
        b_h, hnext, H, /*relu=*/1);
    // out_t = h_next @ W_hq + b_q
    rnn_gemm_bf16_kernel<<<grid_q, 128, 0, stream>>>(
        hnext, H, H, WhqT,
        nullptr, 0, 0, nullptr,
        b_q, out + (size_t)t * B * Q, Q, /*relu=*/0);
    float* tmp = hcur; hcur = hnext; hnext = tmp;
  }

  hipMemcpyAsync(h_out, hcur, (size_t)B * H * sizeof(float),
                 hipMemcpyDeviceToDevice, stream);
}
